// WaterSortGNN_69776038691444
// MI455X (gfx1250) — compile-verified, matrix operand order checked by reading, not run on hardware
//
#include <hip/hip_runtime.h>
#include <hip/hip_bf16.h>

typedef __attribute__((ext_vector_type(16))) _Float16 v16h;
typedef __attribute__((ext_vector_type(8)))  float    v8f;

#define B_    4096
#define T_    16
#define L_    8
#define H_    128
#define NH_   8
#define DFF_  2048
#define LN_EPS 1e-5f

// ---------- WMMA helpers (16x16x32 f16 -> f32) ----------

__device__ __forceinline__ v8f wmma32(v16h a, v16h b, v8f c) {
    return __builtin_amdgcn_wmma_f32_16x16x32_f16(
        /*neg_a=*/false, a, /*neg_b=*/false, b,
        /*c_mod=*/(short)0, c, /*reuse_a=*/false, /*reuse_b=*/false);
}

// A fragment: 16x32 f16 tile from LDS (row-major, leading dim lda), k offset k0.
__device__ __forceinline__ v16h load_a_lds(const _Float16* lds, int lda, int k0) {
    int lane = threadIdx.x & 31;
    int m    = lane & 15;
    int kh   = (lane >> 4) * 8;
    const _Float16* p = lds + m * lda + k0;
    v16h a;
#pragma unroll
    for (int j = 0; j < 8; ++j) {
        a[j]     = p[kh + j];
        a[8 + j] = p[16 + kh + j];
    }
    return a;
}

// B fragment for D = A x B with B[k][n] = W[n][k], W row-major [N,K] f16 in global.
__device__ __forceinline__ v16h load_b_w(const _Float16* W, int ldw, int n0, int k0) {
    int lane = threadIdx.x & 31;
    int n    = n0 + (lane & 15);
    int kh   = (lane >> 4) * 8;
    const _Float16* p = W + (size_t)n * ldw + k0;
    v16h b;
#pragma unroll
    for (int j = 0; j < 8; ++j) {
        b[j]     = p[kh + j];
        b[8 + j] = p[16 + kh + j];
    }
    return b;
}

// Async global->LDS stage of `halves` f16 elements (multiple of 8*256) from gsrc
// into LDS at `dst`. Uses GLOBAL_LOAD_ASYNC_TO_LDS_B128 (ASYNCcnt).
__device__ __forceinline__ void async_stage_f16(_Float16* dst, const _Float16* gsrc,
                                                int halves) {
    int tid = threadIdx.x;
    unsigned lds0 = (unsigned)(uintptr_t)dst;   // low 32 bits = LDS byte offset
    for (int c = tid; c < (halves >> 3); c += 256) {
        unsigned laddr = lds0 + c * 16;
        const _Float16* ga = gsrc + c * 8;
        asm volatile("global_load_async_to_lds_b128 %0, %1, off"
                     :: "v"(laddr), "v"(ga) : "memory");
    }
    asm volatile("s_wait_asynccnt 0x0" ::: "memory");
}

// C/D layout: VGPR r, lanes 0-15 -> M=r, N=lane; lanes 16-31 -> M=r+8, N=lane-16.

// ---------- small kernels ----------

__global__ void f32_to_f16_kernel(const float* __restrict__ src,
                                  _Float16* __restrict__ dst, int n) {
    int i = blockIdx.x * blockDim.x + threadIdx.x;
    if (i < n) dst[i] = (_Float16)src[i];
}

__global__ void embed_mean_kernel(const int* __restrict__ state,
                                  const float* __restrict__ emb,
                                  float* __restrict__ out,
                                  _Float16* __restrict__ out16) {
    int bt = blockIdx.x;
    int h  = threadIdx.x;
    const int* st = state + (size_t)bt * L_;
    float s = 0.f;
#pragma unroll
    for (int l = 0; l < L_; ++l)
        s += emb[(size_t)st[l] * H_ + h];
    s *= (1.0f / L_);
    out[(size_t)bt * H_ + h]   = s;
    out16[(size_t)bt * H_ + h] = (_Float16)s;
}

// agg16[i][h] = sum_{j != i, j<16} x[j][h]
__global__ void gnn_agg_kernel(const float* __restrict__ x, float* __restrict__ agg16) {
    int h = threadIdx.x;
    float s = 0.f;
#pragma unroll
    for (int t = 0; t < 16; ++t) s += x[t * H_ + h];
#pragma unroll
    for (int i = 0; i < 16; ++i) agg16[i * H_ + h] = s - x[i * H_ + h];
}

// ---------- GIN MLP: 64 rows / block ----------
__global__ __launch_bounds__(256)
void gnn_mlp_kernel(const _Float16* __restrict__ x16, const float* __restrict__ agg16,
                    const _Float16* __restrict__ W1h, const float* __restrict__ b1,
                    const _Float16* __restrict__ W2h, const float* __restrict__ b2,
                    float* __restrict__ y, _Float16* __restrict__ y16) {
    __shared__ _Float16 hs[64 * H_];   // 16 KB
    __shared__ _Float16 h2[64 * H_];   // 16 KB
    int tid  = threadIdx.x;
    int row0 = blockIdx.x * 64;

    async_stage_f16(hs, x16 + (size_t)row0 * H_, 64 * H_);
    __syncthreads();
    if (row0 == 0) {   // uniform per block: patch graph-aggregated rows 0..15
        for (int i = tid; i < 16 * H_; i += 256)
            hs[i] = (_Float16)((float)hs[i] + agg16[i]);
        __syncthreads();
    }

    int wave = tid >> 5, lane = tid & 31;
    int n  = wave * 16 + (lane & 15);
    int mb = (lane >> 4) * 8;

    // stage 1
    v8f a1[4] = {{}, {}, {}, {}};
#pragma unroll
    for (int k0 = 0; k0 < H_; k0 += 32) {
        v16h af[4];
#pragma unroll
        for (int mt = 0; mt < 4; ++mt) af[mt] = load_a_lds(hs + mt * 16 * H_, H_, k0);
        v16h bf = load_b_w(W1h, H_, wave * 16, k0);
#pragma unroll
        for (int mt = 0; mt < 4; ++mt) a1[mt] = wmma32(af[mt], bf, a1[mt]);
    }
#pragma unroll
    for (int mt = 0; mt < 4; ++mt)
#pragma unroll
        for (int r = 0; r < 8; ++r) {
            float v = a1[mt][r] + b1[n];
            h2[(mt * 16 + mb + r) * H_ + n] = (_Float16)(v > 0.f ? v : 0.f);
        }
    __syncthreads();

    // stage 2
    v8f a2[4] = {{}, {}, {}, {}};
#pragma unroll
    for (int k0 = 0; k0 < H_; k0 += 32) {
        v16h af[4];
#pragma unroll
        for (int mt = 0; mt < 4; ++mt) af[mt] = load_a_lds(h2 + mt * 16 * H_, H_, k0);
        v16h bf = load_b_w(W2h, H_, wave * 16, k0);
#pragma unroll
        for (int mt = 0; mt < 4; ++mt) a2[mt] = wmma32(af[mt], bf, a2[mt]);
    }
#pragma unroll
    for (int mt = 0; mt < 4; ++mt)
#pragma unroll
        for (int r = 0; r < 8; ++r) {
            float v = a2[mt][r] + b2[n];
            v = v > 0.f ? v : 0.f;
            size_t o = (size_t)(row0 + mt * 16 + mb + r) * H_ + n;
            y[o]   = v;
            y16[o] = (_Float16)v;
        }
}

// ---------- MHA + LN1: 32 rows (2 batches) / block ----------
__global__ __launch_bounds__(256)
void mha_kernel(const float* __restrict__ x, const _Float16* __restrict__ x16,
                const _Float16* __restrict__ wqkvh, const float* __restrict__ bqkv,
                const _Float16* __restrict__ woh,   const float* __restrict__ bo,
                const float* __restrict__ g1, const float* __restrict__ be1,
                float* __restrict__ y, _Float16* __restrict__ y16) {
    __shared__ _Float16 xs[32 * H_];        // 8 KB
    __shared__ _Float16 qkvh[32 * 3 * H_];  // 24 KB
    __shared__ _Float16 ah[32 * H_];        // 8 KB
    __shared__ float    pre[32 * H_];       // 16 KB
    int tid  = threadIdx.x;
    int row0 = blockIdx.x * 32;

    async_stage_f16(xs, x16 + (size_t)row0 * H_, 32 * H_);
    __syncthreads();

    int wave = tid >> 5, lane = tid & 31;
    int mb = (lane >> 4) * 8;

    // QKV: 24 N-tiles, 3 per wave, 2 M-subtiles each
    for (int tt = wave; tt < 24; tt += 8) {
        v8f acc[2] = {{}, {}};
#pragma unroll
        for (int k0 = 0; k0 < H_; k0 += 32) {
            v16h af[2];
#pragma unroll
            for (int mt = 0; mt < 2; ++mt) af[mt] = load_a_lds(xs + mt * 16 * H_, H_, k0);
            v16h bf = load_b_w(wqkvh, H_, tt * 16, k0);
#pragma unroll
            for (int mt = 0; mt < 2; ++mt) acc[mt] = wmma32(af[mt], bf, acc[mt]);
        }
        int n = tt * 16 + (lane & 15);
#pragma unroll
        for (int mt = 0; mt < 2; ++mt)
#pragma unroll
            for (int r = 0; r < 8; ++r)
                qkvh[(mt * 16 + mb + r) * (3 * H_) + n] =
                    (_Float16)(acc[mt][r] + bqkv[n]);
    }
    __syncthreads();

    // attention: 16 tasks (2 batches x 8 heads); wave does task wave and wave+8
#pragma unroll
    for (int tb = 0; tb < 2; ++tb) {
        int task = wave + tb * 8;
        int bl = task >> 3;
        int hd = task & 7;
        if (lane < 16) {
            int m = bl * 16 + lane;
            float qr[16], sc[16];
#pragma unroll
            for (int d = 0; d < 16; ++d)
                qr[d] = (float)qkvh[m * (3 * H_) + hd * 16 + d];
            float mx = -1e30f;
#pragma unroll
            for (int nn = 0; nn < 16; ++nn) {
                int kr = bl * 16 + nn;
                float s = 0.f;
#pragma unroll
                for (int d = 0; d < 16; ++d)
                    s += qr[d] * (float)qkvh[kr * (3 * H_) + H_ + hd * 16 + d];
                s *= 0.25f;   // 1/sqrt(16)
                sc[nn] = s;
                mx = s > mx ? s : mx;
            }
            float sum = 0.f;
#pragma unroll
            for (int nn = 0; nn < 16; ++nn) { sc[nn] = __expf(sc[nn] - mx); sum += sc[nn]; }
            float inv = 1.0f / sum;
#pragma unroll
            for (int d = 0; d < 16; ++d) {
                float o = 0.f;
#pragma unroll
                for (int nn = 0; nn < 16; ++nn)
                    o += sc[nn] * (float)qkvh[(bl * 16 + nn) * (3 * H_) + 2 * H_ + hd * 16 + d];
                ah[m * H_ + hd * 16 + d] = (_Float16)(o * inv);
            }
        }
    }
    __syncthreads();

    // output projection
    {
        v8f acc[2] = {{}, {}};
#pragma unroll
        for (int k0 = 0; k0 < H_; k0 += 32) {
            v16h af[2];
#pragma unroll
            for (int mt = 0; mt < 2; ++mt) af[mt] = load_a_lds(ah + mt * 16 * H_, H_, k0);
            v16h bf = load_b_w(woh, H_, wave * 16, k0);
#pragma unroll
            for (int mt = 0; mt < 2; ++mt) acc[mt] = wmma32(af[mt], bf, acc[mt]);
        }
        int n = wave * 16 + (lane & 15);
#pragma unroll
        for (int mt = 0; mt < 2; ++mt)
#pragma unroll
            for (int r = 0; r < 8; ++r)
                pre[(mt * 16 + mb + r) * H_ + n] = acc[mt][r] + bo[n];
    }
    __syncthreads();

    // residual (from global f32 x) + layernorm; one thread per row
    if (tid < 32) {
        int m = tid;
        const float* xr = x + (size_t)(row0 + m) * H_;
        float mu = 0.f;
        for (int h = 0; h < H_; ++h) {
            float v = pre[m * H_ + h] + xr[h];
            pre[m * H_ + h] = v;
            mu += v;
        }
        mu *= (1.0f / H_);
        float var = 0.f;
        for (int h = 0; h < H_; ++h) { float d = pre[m * H_ + h] - mu; var += d * d; }
        var *= (1.0f / H_);
        float inv = rsqrtf(var + LN_EPS);
        for (int h = 0; h < H_; ++h) {
            float v = (pre[m * H_ + h] - mu) * inv * g1[h] + be1[h];
            size_t o = (size_t)(row0 + m) * H_ + h;
            y[o]   = v;
            y16[o] = (_Float16)v;
        }
    }
}

// ---------- FFN + LN2: 64 rows / block, DFF chunked by 128 ----------
__global__ __launch_bounds__(256)
void ffn_kernel(const float* __restrict__ x, const _Float16* __restrict__ x16,
                const _Float16* __restrict__ w1h, const float* __restrict__ b1,
                const _Float16* __restrict__ w2h, const float* __restrict__ b2,
                const float* __restrict__ g2, const float* __restrict__ be2,
                float* __restrict__ y, _Float16* __restrict__ y16) {
    // 32 KB shared: xs(16K) + hb(16K) during GEMMs, reused as pre(32K) f32 after
    __shared__ __align__(16) char smem[64 * H_ * 2 * 2];
    _Float16* xs = (_Float16*)smem;
    _Float16* hb = (_Float16*)(smem + 64 * H_ * sizeof(_Float16));
    float*    pre = (float*)smem;

    int tid  = threadIdx.x;
    int row0 = blockIdx.x * 64;

    async_stage_f16(xs, x16 + (size_t)row0 * H_, 64 * H_);
    __syncthreads();

    int wave = tid >> 5, lane = tid & 31;
    int n  = wave * 16 + (lane & 15);
    int mb = (lane >> 4) * 8;

    v8f acc[4] = {{}, {}, {}, {}};   // persistent output accumulators
    for (int c = 0; c < DFF_ / 128; ++c) {
        // stage 1: h-chunk = relu(x @ W1^T + b1)
        v8f a1[4] = {{}, {}, {}, {}};
#pragma unroll
        for (int k0 = 0; k0 < H_; k0 += 32) {
            v16h af[4];
#pragma unroll
            for (int mt = 0; mt < 4; ++mt) af[mt] = load_a_lds(xs + mt * 16 * H_, H_, k0);
            v16h bf = load_b_w(w1h, H_, c * 128 + wave * 16, k0);
#pragma unroll
            for (int mt = 0; mt < 4; ++mt) a1[mt] = wmma32(af[mt], bf, a1[mt]);
        }
        int dn = c * 128 + n;
#pragma unroll
        for (int mt = 0; mt < 4; ++mt)
#pragma unroll
            for (int r = 0; r < 8; ++r) {
                float v = a1[mt][r] + b1[dn];
                hb[(mt * 16 + mb + r) * H_ + n] = (_Float16)(v > 0.f ? v : 0.f);
            }
        __syncthreads();
        // stage 2: accumulate h-chunk @ W2^T
#pragma unroll
        for (int k0 = 0; k0 < H_; k0 += 32) {
            v16h af[4];
#pragma unroll
            for (int mt = 0; mt < 4; ++mt) af[mt] = load_a_lds(hb + mt * 16 * H_, H_, k0);
            v16h bf = load_b_w(w2h, DFF_, wave * 16, c * 128 + k0);
#pragma unroll
            for (int mt = 0; mt < 4; ++mt) acc[mt] = wmma32(af[mt], bf, acc[mt]);
        }
        __syncthreads();
    }

    // write pre-LN values (reuses xs/hb space; all GEMM reads are done)
#pragma unroll
    for (int mt = 0; mt < 4; ++mt)
#pragma unroll
        for (int r = 0; r < 8; ++r)
            pre[(mt * 16 + mb + r) * H_ + n] = acc[mt][r] + b2[n];
    __syncthreads();

    // residual + layernorm; one thread per row
    if (tid < 64) {
        int m = tid;
        const float* xr = x + (size_t)(row0 + m) * H_;
        float mu = 0.f;
        for (int h = 0; h < H_; ++h) {
            float v = pre[m * H_ + h] + xr[h];
            pre[m * H_ + h] = v;
            mu += v;
        }
        mu *= (1.0f / H_);
        float var = 0.f;
        for (int h = 0; h < H_; ++h) { float d = pre[m * H_ + h] - mu; var += d * d; }
        var *= (1.0f / H_);
        float inv = rsqrtf(var + LN_EPS);
        for (int h = 0; h < H_; ++h) {
            float v = (pre[m * H_ + h] - mu) * inv * g2[h] + be2[h];
            size_t o = (size_t)(row0 + m) * H_ + h;
            y[o]   = v;
            y16[o] = (_Float16)v;
        }
    }
}

// ---------- pairwise head ----------
__global__ __launch_bounds__(256)
void head_kernel(const float* __restrict__ x, const float* __restrict__ head_w,
                 const float* __restrict__ head_b, float* __restrict__ out) {
    __shared__ float w0[H_], w1s[H_];
    __shared__ float pa[16][16], pc[16][16];
    __shared__ float av[16], cv[16];
    int b = blockIdx.x, tid = threadIdx.x;
    if (tid < H_) { w0[tid] = head_w[tid]; w1s[tid] = head_w[H_ + tid]; }
    __syncthreads();
    int t = tid >> 4, seg = tid & 15;
    const float* row = x + (size_t)(b * 16 + t) * H_ + seg * 8;
    float sa = 0.f, sc = 0.f;
#pragma unroll
    for (int k = 0; k < 8; ++k) {
        float v = row[k];
        sa += v * w0[seg * 8 + k];
        sc += v * w1s[seg * 8 + k];
    }
    pa[t][seg] = sa; pc[t][seg] = sc;
    __syncthreads();
    if (tid < 16) {
        float s1 = 0.f, s2 = 0.f;
#pragma unroll
        for (int s = 0; s < 16; ++s) { s1 += pa[tid][s]; s2 += pc[tid][s]; }
        av[tid] = s1; cv[tid] = s2;
    }
    __syncthreads();
    if (tid < 240) {
        int i = tid / 15, r = tid % 15;
        int j = r + (r >= i ? 1 : 0);
        out[(size_t)b * 240 + tid] = av[i] + cv[j] + head_b[0];
    }
}

// ---------- host launcher ----------

extern "C" void kernel_launch(void* const* d_in, const int* in_sizes, int n_in,
                              void* d_out, int out_size, void* d_ws, size_t ws_size,
                              hipStream_t stream) {
    (void)in_sizes; (void)n_in; (void)out_size; (void)ws_size;
    const int*   state = (const int*)d_in[0];
    const float* emb   = (const float*)d_in[1];
    const float *gW1[3], *gb1[3], *gW2[3], *gb2[3];
    for (int l = 0; l < 3; ++l) {
        int base = 2 + 4 * l;
        gW1[l] = (const float*)d_in[base + 0];
        gb1[l] = (const float*)d_in[base + 1];
        gW2[l] = (const float*)d_in[base + 2];
        gb2[l] = (const float*)d_in[base + 3];
    }
    const float *wqkv[2], *bqkv[2], *wo[2], *bo[2], *w1[2], *b1[2], *w2[2], *b2[2],
                *g1[2], *be1[2], *g2[2], *be2[2];
    for (int l = 0; l < 2; ++l) {
        int base = 14 + 12 * l;
        wqkv[l] = (const float*)d_in[base + 0];  bqkv[l] = (const float*)d_in[base + 1];
        wo[l]   = (const float*)d_in[base + 2];  bo[l]   = (const float*)d_in[base + 3];
        w1[l]   = (const float*)d_in[base + 4];  b1[l]   = (const float*)d_in[base + 5];
        w2[l]   = (const float*)d_in[base + 6];  b2[l]   = (const float*)d_in[base + 7];
        g1[l]   = (const float*)d_in[base + 8];  be1[l]  = (const float*)d_in[base + 9];
        g2[l]   = (const float*)d_in[base + 10]; be2[l]  = (const float*)d_in[base + 11];
    }
    const float* head_w = (const float*)d_in[38];
    const float* head_b = (const float*)d_in[39];

    char* ws = (char*)d_ws;
    const size_t ACT_F32 = (size_t)B_ * T_ * H_ * sizeof(float);     // 32 MB
    const size_t ACT_F16 = (size_t)B_ * T_ * H_ * sizeof(_Float16);  // 16 MB
    float*    bufA   = (float*)ws;
    float*    bufB   = (float*)(ws + ACT_F32);
    _Float16* bufA16 = (_Float16*)(ws + 2 * ACT_F32);
    _Float16* bufB16 = (_Float16*)(ws + 2 * ACT_F32 + ACT_F16);
    float*    agg16  = (float*)(ws + 2 * ACT_F32 + 2 * ACT_F16);
    _Float16* pool   = (_Float16*)(ws + 2 * ACT_F32 + 2 * ACT_F16 + 16384);

    auto conv = [&](const float* src, _Float16* dst, int n) {
        f32_to_f16_kernel<<<(n + 255) / 256, 256, 0, stream>>>(src, dst, n);
    };

    size_t off = 0;
    _Float16 *gW1h[3], *gW2h[3];
    for (int l = 0; l < 3; ++l) {
        gW1h[l] = pool + off; off += H_ * H_;
        gW2h[l] = pool + off; off += H_ * H_;
        conv(gW1[l], gW1h[l], H_ * H_);
        conv(gW2[l], gW2h[l], H_ * H_);
    }
    _Float16 *wqkvh[2], *woh[2], *w1h[2], *w2h[2];
    for (int l = 0; l < 2; ++l) {
        wqkvh[l] = pool + off; off += 3 * H_ * H_;
        woh[l]   = pool + off; off += H_ * H_;
        w1h[l]   = pool + off; off += (size_t)DFF_ * H_;
        w2h[l]   = pool + off; off += (size_t)H_ * DFF_;
        conv(wqkv[l], wqkvh[l], 3 * H_ * H_);
        conv(wo[l],   woh[l],   H_ * H_);
        conv(w1[l],   w1h[l],   DFF_ * H_);
        conv(w2[l],   w2h[l],   H_ * DFF_);
    }

    // 1) embedding + mean over L (writes f32 + f16 shadow)
    embed_mean_kernel<<<B_ * T_, H_, 0, stream>>>(state, emb, bufA, bufA16);

    float*    cur   = bufA;   float*    nxt   = bufB;
    _Float16* cur16 = bufA16; _Float16* nxt16 = bufB16;

    // 2) GIN layers
    for (int l = 0; l < 3; ++l) {
        gnn_agg_kernel<<<1, H_, 0, stream>>>(cur, agg16);
        gnn_mlp_kernel<<<B_ * T_ / 64, 256, 0, stream>>>(
            cur16, agg16, gW1h[l], gb1[l], gW2h[l], gb2[l], nxt, nxt16);
        { float* t = cur; cur = nxt; nxt = t; }
        { _Float16* t = cur16; cur16 = nxt16; nxt16 = t; }
    }

    // 3) transformer layers
    for (int l = 0; l < 2; ++l) {
        mha_kernel<<<B_ / 2, 256, 0, stream>>>(cur, cur16, wqkvh[l], bqkv[l],
                                               woh[l], bo[l], g1[l], be1[l],
                                               nxt, nxt16);
        { float* t = cur; cur = nxt; nxt = t; }
        { _Float16* t = cur16; cur16 = nxt16; nxt16 = t; }
        ffn_kernel<<<B_ * T_ / 64, 256, 0, stream>>>(cur, cur16, w1h[l], b1[l],
                                                     w2h[l], b2[l], g2[l], be2[l],
                                                     nxt, nxt16);
        { float* t = cur; cur = nxt; nxt = t; }
        { _Float16* t = cur16; cur16 = nxt16; nxt16 = t; }
    }

    // 4) pairwise head
    head_kernel<<<B_, 256, 0, stream>>>(cur, head_w, head_b, (float*)d_out);
}
